// MultiHeadAttentionAspectionExtraction_84542136254715
// MI455X (gfx1250) — matrix operand order repaired and checked by reading, not compiled, not run on hardware
//
#include <hip/hip_runtime.h>

#define NB   16
#define NS   1024
#define NE   300
#define NEP  320   // NE padded to a multiple of 32 (bf16, 16B-aligned rows)
#define NH_  4
#define HD   128
#define H2D  256
#define G3   384

typedef __attribute__((ext_vector_type(16))) __bf16 v16bf;
typedef __attribute__((ext_vector_type(8)))  float  v8f;
typedef __attribute__((ext_vector_type(4)))  unsigned tdm_v4u;
typedef __attribute__((ext_vector_type(8)))  int      tdm_v8i;
typedef __attribute__((ext_vector_type(4)))  int      tdm_v4i;

struct u4x2 { uint4 a; uint4 b; };

__device__ __forceinline__ __bf16 f2bf(float f) {
  unsigned u = __builtin_bit_cast(unsigned, f);
  unsigned short r = (unsigned short)((u + 0x7FFFu + ((u >> 16) & 1u)) >> 16);
  return __builtin_bit_cast(__bf16, r);
}
__device__ __forceinline__ v8f zero8() {
  v8f z;
#pragma unroll
  for (int i = 0; i < 8; ++i) z[i] = 0.0f;
  return z;
}
__device__ __forceinline__ v8f wmma_bf16(v16bf a, v16bf b, v8f c) {
  return __builtin_amdgcn_wmma_f32_16x16x32_bf16(false, a, false, b, (short)0, c,
                                                 false, false);
}
// A fragment from row-major [M,K] bf16 tile at p (16B-aligned), row stride ld.
__device__ __forceinline__ v16bf load_a_frag(const __bf16* p, int ld) {
  int lane = threadIdx.x & 31;
  const __bf16* r = p + (lane & 15) * ld + ((lane >> 4) << 4);
  u4x2 t;
  t.a = *(const uint4*)r;         // K {16h..16h+7}
  t.b = *(const uint4*)(r + 8);   // K {16h+8..16h+15}
  return __builtin_bit_cast(v16bf, t);
}
// B fragment for X @ W^T, W row-major [N,K] (16B-aligned rows).
__device__ __forceinline__ v16bf load_bT_frag(const __bf16* p, int ld) {
  int lane = threadIdx.x & 31;
  const __bf16* r = p + (lane & 15) * ld + ((lane >> 4) << 4);
  u4x2 t;
  t.a = *(const uint4*)r;
  t.b = *(const uint4*)(r + 8);
  return __builtin_bit_cast(v16bf, t);
}
__device__ __forceinline__ v16bf load_bT_frag_f32(const float* p, int ld) {
  int lane = threadIdx.x & 31;
  const float* r = p + (lane & 15) * ld + ((lane >> 4) << 4);
  v16bf b;
#pragma unroll
  for (int j = 0; j < 16; ++j) b[j] = f2bf(r[j]);
  return b;
}
// B fragment for X @ W, W row-major [K,N] (column gather, from LDS).
__device__ __forceinline__ v16bf load_b_frag(const __bf16* p, int ld) {
  int lane = threadIdx.x & 31;
  const __bf16* r = p + ((lane >> 4) << 4) * ld + (lane & 15);
  v16bf b;
#pragma unroll
  for (int j = 0; j < 16; ++j) b[j] = r[j * ld];
  return b;
}

// ---------------- K0: one-time fp32 -> bf16 (optionally row-padded) ----------
__global__ __launch_bounds__(256) void cvt_pad_kernel(
    const float* __restrict__ src, __bf16* __restrict__ dst, int src_w,
    int dst_w, int total8) {
  int i = blockIdx.x * 256 + threadIdx.x;
  if (i >= total8) return;
  int e0 = i * 8;
  int row = e0 / dst_w, col = e0 % dst_w;
  const float* s = src + (size_t)row * src_w;
  __bf16* d = dst + (size_t)row * dst_w + col;
#pragma unroll
  for (int j = 0; j < 8; ++j) {
    int k = col + j;
    d[j] = (k < src_w) ? f2bf(s[k]) : f2bf(0.0f);
  }
}

// ---------------- K1: embedding gather + x @ w_ih^T + b_ih (both dirs) -------
__global__ __launch_bounds__(256) void gx_kernel(
    const int* __restrict__ review, const __bf16* __restrict__ emb_bf,
    const __bf16* __restrict__ wih_bf, const float* __restrict__ b_ih_f,
    const float* __restrict__ b_ih_b, float* __restrict__ gx_f,
    float* __restrict__ gx_b) {
  __shared__ __align__(16) __bf16 sA[64][32];
  __shared__ __align__(16) __bf16 sW[G3][32];
  const int dir = blockIdx.y;
  const __bf16* wih = wih_bf + (size_t)dir * G3 * NEP;
  const float* b_ih = dir ? b_ih_b : b_ih_f;
  float* gx = dir ? gx_b : gx_f;
  const int m0 = blockIdx.x * 64;
  const int tid = threadIdx.x, lane = tid & 31, wid = tid >> 5;
  const int mt = wid & 3;          // m-tile 0..3
  const int nt0 = (wid >> 2) * 12; // 12 n-tiles per wave (24 total)
  v8f acc[12];
#pragma unroll
  for (int i = 0; i < 12; ++i) acc[i] = zero8();
  // this thread stages one 16B chunk of one A row each k-step
  const int arow = tid >> 2, aseg = (tid & 3) * 8;
  const int am = m0 + arow;
  const int ab = am & 15, asr = am >> 4;
  const int atime = dir ? (NS - 1 - asr) : asr;
  const long tok = (long)review[ab * NS + atime];
  const __bf16* erow = emb_bf + (size_t)tok * NEP;
  for (int ks = 0; ks < 10; ++ks) {  // K = 300 zero-padded to 320
    const int k0 = ks * 32;
    *(uint4*)&sA[arow][aseg] = *(const uint4*)&erow[k0 + aseg];
    for (int i = tid; i < G3 * 4; i += 256) {
      int rw = i >> 2, cs = (i & 3) * 8;
      *(uint4*)&sW[rw][cs] = *(const uint4*)&wih[(size_t)rw * NEP + k0 + cs];
    }
    __syncthreads();
    v16bf a = load_a_frag(&sA[mt * 16][0], 32);
#pragma unroll
    for (int i = 0; i < 12; ++i) {
      v16bf bf = load_bT_frag(&sW[(nt0 + i) * 16][0], 32);
      acc[i] = wmma_bf16(a, bf, acc[i]);
    }
    __syncthreads();
  }
#pragma unroll
  for (int i = 0; i < 12; ++i) {
    int col = (nt0 + i) * 16 + (lane & 15);
    float bias = b_ih[col];
    int rbase = m0 + mt * 16 + ((lane >> 4) << 3);
#pragma unroll
    for (int j = 0; j < 8; ++j)
      gx[(size_t)(rbase + j) * G3 + col] = acc[i][j] + bias;
  }
}

// ---------------- K2: persistent GRU scan (one block per direction) ----------
__global__ __launch_bounds__(256) void gru_scan_kernel(
    const float* __restrict__ w_hh_f, const float* __restrict__ b_hh_f,
    const float* __restrict__ w_hh_b, const float* __restrict__ b_hh_b,
    const float* __restrict__ gx_f, const float* __restrict__ gx_b,
    __bf16* __restrict__ rh) {
  __shared__ __align__(16) __bf16 sH[NB][HD];
  __shared__ float  fH[NB][HD];
  __shared__ float  sGH[NB][G3];
  const int dir = blockIdx.x;
  const float* w_hh = dir ? w_hh_b : w_hh_f;
  const float* b_hh = dir ? b_hh_b : b_hh_f;
  const float* gx = dir ? gx_b : gx_f;
  const int tid = threadIdx.x, lane = tid & 31, wid = tid >> 5;
  for (int i = tid; i < NB * HD; i += 256) {
    ((float*)fH)[i] = 0.0f;
    ((__bf16*)sH)[i] = f2bf(0.0f);
  }
  // recurrent weights [3H,H]=[N,K] held as B-fragments in VGPRs for all steps
  v16bf wf[3][4];
  float bias[3];
#pragma unroll
  for (int i = 0; i < 3; ++i) {
    int nt = wid * 3 + i;
#pragma unroll
    for (int k = 0; k < 4; ++k)
      wf[i][k] = load_bT_frag_f32(w_hh + (size_t)(nt * 16) * HD + k * 32, HD);
    bias[i] = b_hh[nt * 16 + (lane & 15)];
  }
  __syncthreads();
  for (int s = 0; s < NS; ++s) {
    v16bf a[4];
#pragma unroll
    for (int k = 0; k < 4; ++k) a[k] = load_a_frag(&sH[0][k * 32], HD);
#pragma unroll
    for (int i = 0; i < 3; ++i) {
      v8f acc = zero8();
#pragma unroll
      for (int k = 0; k < 4; ++k) acc = wmma_bf16(a[k], wf[i][k], acc);
      int col = (wid * 3 + i) * 16 + (lane & 15);
      int rb = (lane >> 4) << 3;
#pragma unroll
      for (int j = 0; j < 8; ++j) sGH[rb + j][col] = acc[j] + bias[i];
    }
    __syncthreads();
    const float* gxs = gx + (size_t)s * NB * G3;
    for (int e = tid; e < NB * HD; e += 256) {
      int b = e >> 7, hc = e & 127;
      const float* gr = gxs + b * G3;
      if (s + 1 < NS) __builtin_prefetch(gr + NB * G3, 0, 0);
      float r = 1.0f / (1.0f + __expf(-(gr[hc] + sGH[b][hc])));
      float z = 1.0f / (1.0f + __expf(-(gr[HD + hc] + sGH[b][HD + hc])));
      float n = tanhf(gr[2 * HD + hc] + r * sGH[b][2 * HD + hc]);
      float hn = (1.0f - z) * n + z * fH[b][hc];
      fH[b][hc] = hn;
      sH[b][hc] = f2bf(hn);
      int t = dir ? (NS - 1 - s) : s;
      rh[((size_t)b * NS + t) * H2D + dir * HD + hc] = f2bf(hn);
    }
    __syncthreads();
  }
}

// ---------------- K3: u[b,h] = rh[b] @ W_m[h] --------------------------------
__global__ __launch_bounds__(256) void u_kernel(
    const __bf16* __restrict__ rh, const __bf16* __restrict__ wm_bf,
    __bf16* __restrict__ u) {
  __shared__ __align__(16) __bf16 sW[32][H2D];
  const int b = blockIdx.z, h = blockIdx.y;
  const int m0 = blockIdx.x * 128;
  const int tid = threadIdx.x, lane = tid & 31, wid = tid >> 5;
  v8f acc[16];
#pragma unroll
  for (int i = 0; i < 16; ++i) acc[i] = zero8();
  const __bf16* wm = wm_bf + (size_t)h * H2D * H2D;
  const __bf16* arow = rh + ((size_t)b * NS + m0 + wid * 16) * H2D;
  for (int ks = 0; ks < 8; ++ks) {
    for (int i = tid; i < 1024; i += 256) {  // 32x256 bf16 staged as b128
      int kk = i >> 5, seg = (i & 31) * 8;
      *(uint4*)&sW[kk][seg] =
          *(const uint4*)&wm[(size_t)(ks * 32 + kk) * H2D + seg];
    }
    __syncthreads();
    v16bf a = load_a_frag(arow + ks * 32, H2D);
#pragma unroll
    for (int nt = 0; nt < 16; ++nt) {
      v16bf bf = load_b_frag(&sW[0][nt * 16], H2D);
      acc[nt] = wmma_bf16(a, bf, acc[nt]);
    }
    __syncthreads();
  }
  __bf16* ub = u + (((size_t)b * NH_ + h) * NS) * H2D;
  int rbase = m0 + wid * 16 + ((lane >> 4) << 3);
#pragma unroll
  for (int nt = 0; nt < 16; ++nt) {
    int col = nt * 16 + (lane & 15);
#pragma unroll
    for (int j = 0; j < 8; ++j)
      ub[(size_t)(rbase + j) * H2D + col] = f2bf(acc[nt][j]);
  }
}

// ---------------- K4: flash attention + per-head combine (TDM V staging) -----
__global__ __launch_bounds__(256) void attn_kernel(
    const __bf16* __restrict__ u, const __bf16* __restrict__ rh,
    const float* __restrict__ mask, const float* __restrict__ wt_w,
    float* __restrict__ s_comb) {
  __shared__ float  sS[64][64];
  __shared__ __align__(16) __bf16 sP[64][64];
  __shared__ __align__(16) __bf16 sV[64][H2D];
  __shared__ float rowMax[64], rowSum[64], rowScale[64], kmAdd[64];
  const int b = blockIdx.y;
  const int r0 = blockIdx.x * 64;
  const int tid = threadIdx.x, lane = tid & 31, wid = tid >> 5;
  const int mt = wid & 3;    // row tile
  const int ngrp = wid >> 2; // column group 0/1
  v8f comb[8];
#pragma unroll
  for (int i = 0; i < 8; ++i) comb[i] = zero8();
  const __bf16* rhb = rh + (size_t)b * NS * H2D;
  for (int h = 0; h < NH_; ++h) {
    v8f outh[8];
#pragma unroll
    for (int i = 0; i < 8; ++i) outh[i] = zero8();
    if (tid < 64) { rowMax[tid] = -3.0e38f; rowSum[tid] = 0.0f; }
    const __bf16* ubase = u + (((size_t)b * NH_ + h) * NS + r0 + mt * 16) * H2D;
    for (int t = 0; t < 16; ++t) {
      const int t0 = t * 64;
      // stage V tile (32KB contiguous) via the Tensor Data Mover, wave 0 only
      if (wid == 0) {
        unsigned lds = (unsigned)(uintptr_t)&sV[0][0];
        unsigned long long ga =
            (unsigned long long)(uintptr_t)(rhb + (size_t)t0 * H2D);
        tdm_v4u g0;
        g0[0] = 1u;  // count=1, user descriptor
        g0[1] = lds;
        g0[2] = (unsigned)ga;
        g0[3] = (unsigned)((ga >> 32) & 0x1FFFFFFull) | (2u << 30);  // type=2
        tdm_v8i g1;
        g1[0] = 3 << 16;             // data_size = 8B units
        g1[1] = (int)(4096u << 16);  // tensor_dim0 = 4096 (lo16)
        g1[2] = 1 << 16;             // tensor_dim0 hi=0 | tensor_dim1 = 1
        g1[3] = (int)(4096u << 16);  // tile_dim0 = 4096 -> 32KB 1-D tile
        g1[4] = 0; g1[5] = 4096; g1[6] = 0; g1[7] = 0;  // dim0_stride=4096
        tdm_v4i gz4;
        gz4[0] = 0; gz4[1] = 0; gz4[2] = 0; gz4[3] = 0;
        tdm_v8i gz8;
#pragma unroll
        for (int q = 0; q < 8; ++q) gz8[q] = 0;
        __builtin_amdgcn_tensor_load_to_lds(g0, g1, gz4, gz4, gz8, 0);
      }
      if (tid < 64)
        kmAdd[tid] = (mask[(size_t)b * NS + t0 + tid] - 1.0f) * 1.0e11f;
      // scores: rows [mt*16,+16), cols [ngrp*32,+32) of this 64x64 tile; K=256
      v8f sc[2];
      sc[0] = zero8(); sc[1] = zero8();
#pragma unroll
      for (int ks = 0; ks < 8; ++ks) {
        v16bf a = load_a_frag(ubase + ks * 32, H2D);
#pragma unroll
        for (int ci = 0; ci < 2; ++ci) {
          int ct = ngrp * 2 + ci;
          v16bf bf =
              load_bT_frag(rhb + (size_t)(t0 + ct * 16) * H2D + ks * 32, H2D);
          sc[ci] = wmma_bf16(a, bf, sc[ci]);
        }
      }
#pragma unroll
      for (int ci = 0; ci < 2; ++ci) {
        int col = (ngrp * 2 + ci) * 16 + (lane & 15);
        int rb = mt * 16 + ((lane >> 4) << 3);
#pragma unroll
        for (int j = 0; j < 8; ++j) sS[rb + j][col] = sc[ci][j];
      }
      __builtin_amdgcn_s_wait_tensorcnt(0);  // V tile landed (no-op for w1..7)
      __syncthreads();
      if (tid < 64) {  // online softmax, one thread per row
        float mOld = rowMax[tid], mNew = mOld;
        for (int c = 0; c < 64; ++c)
          mNew = fmaxf(mNew, sS[tid][c] + kmAdd[c]);
        float f = __expf(mOld - mNew), sum = 0.0f;
        for (int c = 0; c < 64; ++c) {
          float p = __expf(sS[tid][c] + kmAdd[c] - mNew);
          sP[tid][c] = f2bf(p);
          sum += p;
        }
        rowSum[tid] = rowSum[tid] * f + sum;
        rowMax[tid] = mNew;
        rowScale[tid] = f;
      }
      __syncthreads();
      // rescale accumulators, then out += P @ V
      int rb = mt * 16 + ((lane >> 4) << 3);
      float fs[8];
#pragma unroll
      for (int j = 0; j < 8; ++j) fs[j] = rowScale[rb + j];
#pragma unroll
      for (int i = 0; i < 8; ++i)
#pragma unroll
        for (int j = 0; j < 8; ++j) outh[i][j] *= fs[j];
#pragma unroll
      for (int ks = 0; ks < 2; ++ks) {
        v16bf a = load_a_frag(&sP[mt * 16][ks * 32], 64);
#pragma unroll
        for (int i = 0; i < 8; ++i) {
          v16bf bf = load_b_frag(&sV[ks * 32][(ngrp * 8 + i) * 16], H2D);
          outh[i] = wmma_bf16(a, bf, outh[i]);
        }
      }
      __syncthreads();
    }
    // fold head: comb += wt_w[h] * outh / rowSum
    float w = wt_w[h];
    int rb = mt * 16 + ((lane >> 4) << 3);
    float inv[8];
#pragma unroll
    for (int j = 0; j < 8; ++j) inv[j] = w / rowSum[rb + j];
#pragma unroll
    for (int i = 0; i < 8; ++i)
#pragma unroll
      for (int j = 0; j < 8; ++j) comb[i][j] += outh[i][j] * inv[j];
    __syncthreads();
  }
  int rb = r0 + mt * 16 + ((lane >> 4) << 3);
#pragma unroll
  for (int i = 0; i < 8; ++i) {
    int col = (ngrp * 8 + i) * 16 + (lane & 15);
#pragma unroll
    for (int j = 0; j < 8; ++j)
      s_comb[((size_t)b * NS + rb + j) * H2D + col] = comb[i][j];
  }
}

// ---------------- K5: 256->3 projection + log_softmax + mask -----------------
__global__ __launch_bounds__(256) void out_kernel(
    const float* __restrict__ s_comb, const float* __restrict__ wt_b,
    const float* __restrict__ wr_w, const float* __restrict__ wr_b,
    const float* __restrict__ mask, float* __restrict__ out) {
  const int tid = threadIdx.x, lane = tid & 31, wid = tid >> 5;
  const int row = blockIdx.x * 8 + wid;  // in [0, B*S)
  const float tb = wt_b[0];
  float a0 = 0.f, a1 = 0.f, a2 = 0.f;
  const float* src = s_comb + (size_t)row * H2D;
  for (int c = lane; c < H2D; c += 32) {
    float v = src[c] + tb;
    a0 += v * wr_w[c];
    a1 += v * wr_w[H2D + c];
    a2 += v * wr_w[2 * H2D + c];
  }
#pragma unroll
  for (int off = 16; off > 0; off >>= 1) {
    a0 += __shfl_xor(a0, off, 32);
    a1 += __shfl_xor(a1, off, 32);
    a2 += __shfl_xor(a2, off, 32);
  }
  if (lane == 0) {
    float v0 = a0 + wr_b[0], v1 = a1 + wr_b[1], v2 = a2 + wr_b[2];
    float m = fmaxf(v0, fmaxf(v1, v2));
    float lse = m + logf(__expf(v0 - m) + __expf(v1 - m) + __expf(v2 - m));
    float mv = mask[row];
    out[row * 3 + 0] = (v0 - lse) * mv;
    out[row * 3 + 1] = (v1 - lse) * mv;
    out[row * 3 + 2] = (v2 - lse) * mv;
  }
}

extern "C" void kernel_launch(void* const* d_in, const int* in_sizes, int n_in,
                              void* d_out, int out_size, void* d_ws,
                              size_t ws_size, hipStream_t stream) {
  (void)in_sizes; (void)n_in; (void)out_size; (void)ws_size;
  const int*   review = (const int*)d_in[0];
  const float* mask   = (const float*)d_in[2];
  const float* emb    = (const float*)d_in[3];
  const float* w_ih_f = (const float*)d_in[4];
  const float* w_hh_f = (const float*)d_in[5];
  const float* b_ih_f = (const float*)d_in[6];
  const float* b_hh_f = (const float*)d_in[7];
  const float* w_ih_b = (const float*)d_in[8];
  const float* w_hh_b = (const float*)d_in[9];
  const float* b_ih_b = (const float*)d_in[10];
  const float* b_hh_b = (const float*)d_in[11];
  const float* weight_m = (const float*)d_in[12];
  const float* wt_w = (const float*)d_in[13];
  const float* wt_b = (const float*)d_in[14];
  const float* wr_w = (const float*)d_in[15];
  const float* wr_b = (const float*)d_in[16];

  char* ws = (char*)d_ws;
  float* gx_f = (float*)ws;     ws += (size_t)NB * NS * G3 * 4;
  float* gx_b = (float*)ws;     ws += (size_t)NB * NS * G3 * 4;
  __bf16* rh = (__bf16*)ws;     ws += (size_t)NB * NS * H2D * 2;
  __bf16* u  = (__bf16*)ws;     ws += (size_t)NB * NH_ * NS * H2D * 2;
  float* s_comb = (float*)ws;   ws += (size_t)NB * NS * H2D * 4;
  __bf16* emb_bf = (__bf16*)ws; ws += (size_t)32000 * NEP * 2;
  __bf16* wih_bf = (__bf16*)ws; ws += (size_t)2 * G3 * NEP * 2;
  __bf16* wm_bf  = (__bf16*)ws;

  cvt_pad_kernel<<<dim3(32000 * (NEP / 8) / 256), 256, 0, stream>>>(
      emb, emb_bf, NE, NEP, 32000 * (NEP / 8));
  cvt_pad_kernel<<<dim3(60), 256, 0, stream>>>(w_ih_f, wih_bf, NE, NEP,
                                               G3 * (NEP / 8));
  cvt_pad_kernel<<<dim3(60), 256, 0, stream>>>(
      w_ih_b, wih_bf + (size_t)G3 * NEP, NE, NEP, G3 * (NEP / 8));
  cvt_pad_kernel<<<dim3(128), 256, 0, stream>>>(
      weight_m, wm_bf, H2D, H2D, NH_ * H2D * (H2D / 8));

  gx_kernel<<<dim3((NB * NS) / 64, 2), 256, 0, stream>>>(
      review, emb_bf, wih_bf, b_ih_f, b_ih_b, gx_f, gx_b);
  gru_scan_kernel<<<dim3(2), 256, 0, stream>>>(w_hh_f, b_hh_f, w_hh_b, b_hh_b,
                                               gx_f, gx_b, rh);
  u_kernel<<<dim3(NS / 128, NH_, NB), 256, 0, stream>>>(rh, wm_bf, u);
  attn_kernel<<<dim3(NS / 64, NB), 256, 0, stream>>>(u, rh, mask, wt_w, s_comb);
  out_kernel<<<dim3((NB * NS) / 8), 256, 0, stream>>>(s_comb, wt_b, wr_w, wr_b,
                                                      mask, (float*)d_out);
}
